// StyledConv_29308856828611
// MI455X (gfx1250) — matrix operand order repaired and checked by reading, not compiled, not run on hardware
//
#include <hip/hip_runtime.h>

// ---------------------------------------------------------------------------
// StyleGAN2 modulated conv (non-fused) for MI455X / gfx1250.
// B=8, CIN=COUT=128, H=W=256, K=3, SD=512.
// Implicit-GEMM with V_WMMA_F32_16X16X32_BF16, f32 accumulate.
// Round 3: tap-outer loop with immediate-offset DS loads, grouped B-fragment
// loads, branchless A double-buffer. Keeps VGPRs < 256.
// ---------------------------------------------------------------------------

typedef __attribute__((ext_vector_type(16))) __bf16 v16bf;
typedef __attribute__((ext_vector_type(8)))  float  v8f;

#define NB   8
#define CIN  128
#define COUT 128
#define HH   256
#define WW   256
#define SD   512
#define KKT  9             // 3x3 taps
#define KDIM (CIN * KKT)   // 1152
#define NCHUNK (KDIM / 32) // 36

#define INV_SQRT_SD   0.04419417382415922f   // 1/sqrt(512)
#define WEIGHT_GAIN   0.029462782549439483f  // 1/sqrt(128*9)
#define SQRT2         1.4142135623730951f

__device__ __forceinline__ unsigned short f32_to_bf16_rne(float f) {
  union { float f; unsigned u; } c; c.f = f;
  unsigned u = c.u;
  unsigned r = u + 0x7FFFu + ((u >> 16) & 1u);   // round-to-nearest-even
  return (unsigned short)(r >> 16);
}

// ---------------------------------------------------------------------------
// K1: s[b,i] = (style[b,:] . mod_w[i,:]) / sqrt(SD) + mod_b[i], * weight_gain
// ---------------------------------------------------------------------------
__global__ __launch_bounds__(256)
void k_style(const float* __restrict__ style, const float* __restrict__ mod_w,
             const float* __restrict__ mod_b, float* __restrict__ s_out) {
  int tid = blockIdx.x * 256 + threadIdx.x;     // 1024 threads total
  int b = tid >> 7, i = tid & 127;
  const float* st = style + b * SD;
  const float* mw = mod_w + i * SD;
  float acc = 0.f;
  #pragma unroll 8
  for (int d = 0; d < SD; ++d) acc = fmaf(st[d], mw[d], acc);
  float s = fmaf(acc, INV_SQRT_SD, mod_b[i]);
  s_out[tid] = s * WEIGHT_GAIN;
}

// ---------------------------------------------------------------------------
// K2: dcoefs[b,o] = rsqrt( sum_i (sum_t w[o,i,t]^2) * s[b,i]^2 + eps )
// ---------------------------------------------------------------------------
__global__ __launch_bounds__(256)
void k_dcoef(const float* __restrict__ weight, const float* __restrict__ s,
             float* __restrict__ dcoefs) {
  int tid = blockIdx.x * 256 + threadIdx.x;     // 1024 threads total
  int b = tid >> 7, o = tid & 127;
  const float* srow = s + b * CIN;
  float acc = 0.f;
  for (int i = 0; i < CIN; ++i) {
    const float* w = weight + (o * CIN + i) * KKT;
    float ws = 0.f;
    #pragma unroll
    for (int t = 0; t < KKT; ++t) ws = fmaf(w[t], w[t], ws);
    float si = srow[i];
    acc = fmaf(ws, si * si, acc);
  }
  dcoefs[tid] = rsqrtf(acc + 1e-8f);
}

// ---------------------------------------------------------------------------
// K3: pack weights into WMMA 16x32 bf16 A-fragment layout.
//   apk[(coutTile(8) * 36 + chunk)*32 + lane][16 bf16]   (32 B per lane)
//   K ordering: k = tap*128 + cin  (chunk = 32 consecutive cin of one tap)
//   lane layout used here (contiguous K per lane half; B uses the same order):
//     lanes 0-15 : M=lane,    K=e
//     lanes 16-31: M=lane-16, K=e+16
// ---------------------------------------------------------------------------
__global__ __launch_bounds__(256)
void k_pack(const float* __restrict__ weight, unsigned short* __restrict__ apk) {
  int idx = blockIdx.x * 256 + threadIdx.x;     // 8*36*32*16 = 147456
  int e    = idx & 15;
  int lane = (idx >> 4) & 31;
  int q    = (idx >> 9) % NCHUNK;
  int t    = idx / (16 * 32 * NCHUNK);
  int klocal = (lane < 16) ? e : e + 16;
  int kk  = q * 32 + klocal;
  int tap = kk >> 7;          // kk / 128
  int cin = kk & 127;
  int m   = t * 16 + (lane & 15);
  apk[idx] = f32_to_bf16_rne(weight[(m * CIN + cin) * KKT + tap]);
}

// ---------------------------------------------------------------------------
// K4: main conv. Block = 256 threads (8 wave32).
//   Output tile: 128 cout x (2 rows x 32 cols). Wave w -> cout tile w (16).
//   Each wave: 4 N-tiles of 16 pixels -> 4 f32 16x16 accumulators.
//   LDS: modulated bf16 input halo xs[4 rows][34 cols][136 cin-padded].
//   Loop: 9 taps rolled; 4 chunks x 4 tiles unrolled inside with immediate
//   DS offsets; A fragments double-buffered one tap ahead (branchless).
// ---------------------------------------------------------------------------
__global__ __launch_bounds__(256)
void k_conv(const float* __restrict__ x, const float* __restrict__ noise,
            const float* __restrict__ s, const float* __restrict__ dcoefs,
            const float* __restrict__ bias, const float* __restrict__ nw_p,
            const unsigned short* __restrict__ apk, float* __restrict__ out) {
  int bx    = blockIdx.x;                 // 8 xtiles * 128 ytiles * 8 batch
  int tileX = bx & 7;
  int tileY = (bx >> 3) & 127;
  int b     = bx >> 10;
  int x0    = tileX * 32;
  int y0    = tileY * 2;

  __shared__ __align__(16) unsigned short xs[4][34][136];  // 36992 B
  __shared__ float sS[CIN], sDc[COUT], sB[COUT];

  int tid = threadIdx.x;
  if (tid < 128) {
    sS[tid]  = s[b * CIN + tid];
    sDc[tid] = dcoefs[b * COUT + tid];
    sB[tid]  = bias[tid];
  }
  __syncthreads();

  // ---- stage modulated input halo as bf16 (cin-contiguous, padded stride) --
  // iterate (ry 0..3, cinPair 0..63, col 0..33) = 8704 = 34 * 256 (uniform)
  for (int i = tid; i < 4 * 64 * 34; i += 256) {
    int col = i % 34;
    int cp  = (i / 34) & 63;
    int ry  = i / (34 * 64);
    int c0  = cp * 2;
    int gy  = y0 + ry - 1;
    int gx  = x0 + col - 1;
    float v0 = 0.f, v1 = 0.f;
    if ((unsigned)gy < (unsigned)HH && (unsigned)gx < (unsigned)WW) {
      const float* px = x + (((size_t)b * CIN + c0) * HH + gy) * WW + gx;
      v0 = px[0]       * sS[c0];
      v1 = px[HH * WW] * sS[c0 + 1];
    }
    unsigned pk = (unsigned)f32_to_bf16_rne(v0) |
                  ((unsigned)f32_to_bf16_rne(v1) << 16);
    *(unsigned*)&xs[ry][col][c0] = pk;
  }
  __syncthreads();

  int lane = tid & 31;
  int wv   = tid >> 5;        // cout tile 0..7
  int n    = lane & 15;       // N (pixel within tile)
  int hi   = lane >> 4;       // lane half (selects K half / M half)

  v8f acc[4] = {};            // 4 N-tiles, f32 16x16 accumulators

  // A: 32 B per (chunk,lane); chunk stride = 1024 B from a fixed lane base.
  const char* Abase =
      (const char*)(apk + ((size_t)(wv * NCHUNK) * 32 + lane) * 16);

  // B: lane-dependent LDS byte base; per-tap scalar add; rest are immediates.
  //   byte addr = laneB + tapOff + pOff{0,4352,9248,13600} + cc*64 (+16)
  const char* laneB = (const char*)xs + (n * 136 + hi * 16) * 2;

  union AU { uint4 u[2]; v16bf v; };
  AU aN0, aN1, aN2, aN3;

  // prefetch tap 0 A fragments
  aN0.u[0] = *(const uint4*)(Abase);          aN0.u[1] = *(const uint4*)(Abase + 16);
  aN1.u[0] = *(const uint4*)(Abase + 1024);   aN1.u[1] = *(const uint4*)(Abase + 1040);
  aN2.u[0] = *(const uint4*)(Abase + 2048);   aN2.u[1] = *(const uint4*)(Abase + 2064);
  aN3.u[0] = *(const uint4*)(Abase + 3072);   aN3.u[1] = *(const uint4*)(Abase + 3088);

  #pragma unroll 1
  for (int tap = 0; tap < KKT; ++tap) {
    AU a0 = aN0, a1 = aN1, a2 = aN2, a3 = aN3;

    // branchless prefetch of next tap (clamped -> uniform select, no branch)
    int tapn = (tap + 1 < KKT) ? tap + 1 : tap;
    const char* An = Abase + (size_t)tapn * 4096;
    aN0.u[0] = *(const uint4*)(An);          aN0.u[1] = *(const uint4*)(An + 16);
    aN1.u[0] = *(const uint4*)(An + 1024);   aN1.u[1] = *(const uint4*)(An + 1040);
    aN2.u[0] = *(const uint4*)(An + 2048);   aN2.u[1] = *(const uint4*)(An + 2064);
    aN3.u[0] = *(const uint4*)(An + 3072);   aN3.u[1] = *(const uint4*)(An + 3088);

    int ky = tap / 3;
    int kx = tap - ky * 3;
    const char* Bt = laneB + (ky * 34 + kx) * 272;   // one scalar add per tap

    #pragma unroll
    for (int cc = 0; cc < 4; ++cc) {
      const v16bf a = (cc == 0) ? a0.v : (cc == 1) ? a1.v
                    : (cc == 2) ? a2.v : a3.v;
      // load all 4 B fragments of this chunk, then 4 independent WMMAs
      v16bf b0 = *(const v16bf*)(Bt + cc * 64);
      v16bf b1 = *(const v16bf*)(Bt + cc * 64 + 4352);   // pcol=16
      v16bf b2 = *(const v16bf*)(Bt + cc * 64 + 9248);   // prow=1
      v16bf b3 = *(const v16bf*)(Bt + cc * 64 + 13600);  // prow=1,pcol=16
      acc[0] = __builtin_amdgcn_wmma_f32_16x16x32_bf16(
          false, a, false, b0, (short)0, acc[0], false, false);
      acc[1] = __builtin_amdgcn_wmma_f32_16x16x32_bf16(
          false, a, false, b1, (short)0, acc[1], false, false);
      acc[2] = __builtin_amdgcn_wmma_f32_16x16x32_bf16(
          false, a, false, b2, (short)0, acc[2], false, false);
      acc[3] = __builtin_amdgcn_wmma_f32_16x16x32_bf16(
          false, a, false, b3, (short)0, acc[3], false, false);
    }
  }

  // ---- epilogue: demod * acc + nw*noise + bias, lrelu(0.2) * sqrt(2) ------
  float nwv = nw_p[0];
  #pragma unroll
  for (int p = 0; p < 4; ++p) {
    int prow = p >> 1;
    int pcol = (p & 1) << 4;
    int gy   = y0 + prow;
    int gx   = x0 + pcol + n;
    float nz = nwv * noise[((size_t)b * HH + gy) * WW + gx];
    #pragma unroll
    for (int r = 0; r < 8; ++r) {
      int co  = (wv << 4) + r + (hi << 3);   // C/D layout: VGPR r -> M=r / r+8
      float v = fmaf(acc[p][r], sDc[co], nz) + sB[co];
      v = (v > 0.f ? v : 0.2f * v) * SQRT2;
      out[(((size_t)b * COUT + co) * HH + gy) * WW + gx] = v;
    }
  }
}

// ---------------------------------------------------------------------------
extern "C" void kernel_launch(void* const* d_in, const int* in_sizes, int n_in,
                              void* d_out, int out_size, void* d_ws, size_t ws_size,
                              hipStream_t stream) {
  const float* x      = (const float*)d_in[0];
  const float* style  = (const float*)d_in[1];
  const float* noise  = (const float*)d_in[2];
  const float* weight = (const float*)d_in[3];
  const float* mod_w  = (const float*)d_in[4];
  const float* mod_b  = (const float*)d_in[5];
  const float* nw     = (const float*)d_in[6];
  const float* bias   = (const float*)d_in[7];
  float* out = (float*)d_out;

  char* ws = (char*)d_ws;
  float* s_ws            = (float*)(ws);                 // 1024 f32
  float* dc_ws           = (float*)(ws + 4096);          // 1024 f32
  unsigned short* apk    = (unsigned short*)(ws + 8192); // 147456 u16 (288 KB)

  k_style<<<4,    256, 0, stream>>>(style, mod_w, mod_b, s_ws);
  k_dcoef<<<4,    256, 0, stream>>>(weight, s_ws, dc_ws);
  k_pack <<<576,  256, 0, stream>>>(weight, apk);
  k_conv <<<8192, 256, 0, stream>>>(x, noise, s_ws, dc_ws, bias, nw, apk, out);
}
